// MultiHeadSelfAttention_30734785970866
// MI455X (gfx1250) — matrix operand order repaired and checked by reading, not compiled, not run on hardware
//
#include <hip/hip_runtime.h>

typedef __bf16 bf16;
typedef __attribute__((ext_vector_type(16))) __bf16 v16bf;
typedef __attribute__((ext_vector_type(8)))  __bf16 bf16x8;
typedef __attribute__((ext_vector_type(4)))  __bf16 bf16x4;
typedef __attribute__((ext_vector_type(8)))  float   v8f;

#define E_DIM 1024
#define NHEAD 16
#define HDIM  64
#define VOC   32000
#define BATCH 2
#define SEQ   2048
#define MROWS (BATCH*SEQ)   // 4096

// ---- WMMA helper: D = A(16x32 bf16) * B(32x16 bf16) + C(f32) ----
static __device__ __forceinline__ v8f wmma_bf16(v16bf a, v16bf b, v8f c) {
  return __builtin_amdgcn_wmma_f32_16x16x32_bf16(false, a, false, b, (short)0, c, false, false);
}

// Build a 16-element bf16 fragment from two 16-byte chunks (-> two b128 loads)
static __device__ __forceinline__ v16bf ld_frag2(const bf16* p0, const bf16* p1) {
  bf16x8 lo = *(const bf16x8*)p0;
  bf16x8 hi = *(const bf16x8*)p1;
  v16bf r;
#pragma unroll
  for (int i = 0; i < 8; ++i) { r[i] = lo[i]; r[i + 8] = hi[i]; }
  return r;
}

// ---- CDNA5 async global->LDS DMA (ASYNCcnt-tracked), CDNA5 ISA ----
// VDST = per-lane LDS byte offset (low 32 bits of generic pointer), VADDR = 64-bit global addr.
static __device__ __forceinline__ void async_load_b128(void* lds_ptr, const void* gaddr) {
  unsigned lds_off = (unsigned)(unsigned long long)lds_ptr;
  asm volatile("global_load_async_to_lds_b128 %0, %1, off"
               :: "v"(lds_off), "v"((unsigned long long)gaddr)
               : "memory");
}
static __device__ __forceinline__ void wait_async0() {
  asm volatile("s_wait_asynccnt 0" ::: "memory");
}

// ---------------- elementwise cast f32 -> bf16 (vectorized) ----------------
__global__ void cast_f32_bf16(const float* __restrict__ src, bf16* __restrict__ dst, int n4) {
  int i = blockIdx.x * blockDim.x + threadIdx.x;
  if (i >= n4) return;
  float4 f = ((const float4*)src)[i];
  bf16x4 o = { (bf16)f.x, (bf16)f.y, (bf16)f.z, (bf16)f.w };
  ((bf16x4*)dst)[i] = o;
}

// ------------- LDS-tiled transpose + cast: src[K][N] f32 -> dst[N][K] bf16 -------------
__global__ void transpose_cast(const float* __restrict__ src, bf16* __restrict__ dst,
                               int K, int N) {
  __shared__ float tile[32][33];
  int tx = threadIdx.x, ty = threadIdx.y;
  int n0 = blockIdx.x * 32, k0 = blockIdx.y * 32;
#pragma unroll
  for (int i = 0; i < 32; i += 8)
    tile[ty + i][tx] = src[(size_t)(k0 + ty + i) * N + n0 + tx];
  __syncthreads();
#pragma unroll
  for (int i = 0; i < 32; i += 8)
    dst[(size_t)(n0 + ty + i) * K + k0 + tx] = (bf16)tile[tx][ty + i];
}

// ------------- fused QKV projection: out = x @ W + b, scattered into head layouts -------------
// z=0: Q -> [b,h,s,d]   z=1: K -> [b,h,s,d]   z=2: V -> [b,h,d,s] (transposed for PV WMMA B-frags)
__global__ __launch_bounds__(128) void qkv_gemm(
    const bf16* __restrict__ xb,
    const bf16* __restrict__ wqt, const bf16* __restrict__ wkt, const bf16* __restrict__ wvt,
    const float* __restrict__ bq, const float* __restrict__ bk, const float* __restrict__ bv,
    bf16* __restrict__ Qb, bf16* __restrict__ Kb, bf16* __restrict__ Vt) {
  const int lane = threadIdx.x & 31;
  const int wave = threadIdx.x >> 5;
  const int half = lane >> 4;
  const int nc   = lane & 15;
  const int n0 = blockIdx.x * 16;
  const int m0 = blockIdx.y * 64 + wave * 16;
  const int z  = blockIdx.z;
  const bf16*  wt   = (z == 0) ? wqt : (z == 1) ? wkt : wvt;
  const float* bias = (z == 0) ? bq  : (z == 1) ? bk  : bv;

  const bf16* arow = xb + (size_t)(m0 + nc) * E_DIM;
  const bf16* bcol = wt + (size_t)(n0 + nc) * E_DIM + 8 * half;

  v8f acc = {};
#pragma unroll 2
  for (int k = 0; k < E_DIM; k += 32) {
    v16bf a = ld_frag2(arow + k + half * 8, arow + k + 16 + half * 8);
    v16bf b = ld_frag2(bcol + k,            bcol + k + 16);
    acc = wmma_bf16(a, b, acc);
  }
  const int   col = n0 + nc;
  const float bi  = bias[col];
  const int h = col >> 6, d = col & 63;
#pragma unroll
  for (int r = 0; r < 8; ++r) {
    int row = m0 + r + 8 * half;
    int bb = row >> 11, s = row & (SEQ - 1);
    float v = acc[r] + bi;
    if (z == 2)
      Vt[((size_t)((bb * NHEAD + h) * HDIM + d)) * SEQ + s] = (bf16)v;
    else {
      bf16* dst = (z == 0) ? Qb : Kb;
      dst[((size_t)((bb * NHEAD + h) * SEQ + s)) * HDIM + d] = (bf16)v;
    }
  }
}

// ------------- flash attention: block = (b,h,64 q rows); K/V staged via async DMA -------------
__global__ __launch_bounds__(128) void flash_attn(
    const bf16* __restrict__ Qb, const bf16* __restrict__ Kb,
    const bf16* __restrict__ Vt, bf16* __restrict__ Ob) {
  const int lane = threadIdx.x & 31;
  const int wave = threadIdx.x >> 5;
  const int half = lane >> 4;
  const int nc   = lane & 15;
  const int bh = blockIdx.x >> 5;                    // SEQ/64 == 32 q-blocks per (b,h)
  const int q0 = (blockIdx.x & 31) * 64 + wave * 16;
  const int bb = bh >> 4, h = bh & 15;

  __shared__ __align__(16) bf16 kbuf[32 * HDIM];     // K tile  [32 keys][64 d]   4KB
  __shared__ __align__(16) bf16 vbuf[HDIM * 32];     // V tile  [64 d][32 keys]   4KB
  __shared__ __align__(16) bf16 pst[4][16 * 32];     // per-wave P staging        4KB
  bf16* lds = pst[wave];

  const bf16* qrow = Qb + ((size_t)bh * SEQ + q0 + nc) * HDIM;
  v16bf aq0 = ld_frag2(qrow + half * 8,      qrow + 16 + half * 8);  // d 0..31
  v16bf aq1 = ld_frag2(qrow + 32 + half * 8, qrow + 48 + half * 8);  // d 32..63

  float m_run[8], l_run[8];
#pragma unroll
  for (int r = 0; r < 8; ++r) { m_run[r] = -3.0e38f; l_run[r] = 0.f; }
  v8f acc[4] = {};

  const bf16* kbase = Kb + (size_t)bh * SEQ * HDIM;
  const bf16* vbase = Vt + (size_t)bh * HDIM * SEQ;

  for (int kb = 0; kb < SEQ; kb += 32) {
    // ---- stage 8KB of K/V into LDS: 512 x b128 async copies over 128 threads ----
#pragma unroll
    for (int i = 0; i < 2; ++i) {
      int c = threadIdx.x * 2 + i;
      int krow = c >> 3, kpart = c & 7;              // 32 rows x 128B
      async_load_b128(&kbuf[krow * HDIM + kpart * 8],
                      kbase + (size_t)(kb + krow) * HDIM + kpart * 8);
      int vrow = c >> 2, vpart = c & 3;              // 64 rows x 64B
      async_load_b128(&vbuf[vrow * 32 + vpart * 8],
                      vbase + (size_t)vrow * SEQ + kb + vpart * 8);
    }
    wait_async0();
    __syncthreads();

    // --- scores: two 16x16 tiles over 32 keys, K-dim = head dim (2 slices) ---
    const bf16* kc0 = kbuf + nc * HDIM + 8 * half;
    const bf16* kc1 = kbuf + (nc + 16) * HDIM + 8 * half;
    v16bf b00 = ld_frag2(kc0,      kc0 + 16);
    v16bf b01 = ld_frag2(kc0 + 32, kc0 + 48);
    v16bf b10 = ld_frag2(kc1,      kc1 + 16);
    v16bf b11 = ld_frag2(kc1 + 32, kc1 + 48);
    v8f s0 = {}, s1 = {};
    s0 = wmma_bf16(aq0, b00, s0); s0 = wmma_bf16(aq1, b01, s0);
    s1 = wmma_bf16(aq0, b10, s1); s1 = wmma_bf16(aq1, b11, s1);

    // --- online softmax (row stats match C-fragment layout) ---
#pragma unroll
    for (int r = 0; r < 8; ++r) {
      float sa = s0[r] * 0.125f, sb = s1[r] * 0.125f;   // 1/sqrt(64)
      float mx = fmaxf(sa, sb);
      mx = fmaxf(mx, __shfl_xor(mx, 1, 32));
      mx = fmaxf(mx, __shfl_xor(mx, 2, 32));
      mx = fmaxf(mx, __shfl_xor(mx, 4, 32));
      mx = fmaxf(mx, __shfl_xor(mx, 8, 32));
      float mnew = fmaxf(m_run[r], mx);
      float sf = __expf(m_run[r] - mnew);
      m_run[r] = mnew;
      float p0 = __expf(sa - mnew);
      float p1 = __expf(sb - mnew);
      float ps = p0 + p1;
      ps += __shfl_xor(ps, 1, 32);
      ps += __shfl_xor(ps, 2, 32);
      ps += __shfl_xor(ps, 4, 32);
      ps += __shfl_xor(ps, 8, 32);
      l_run[r] = l_run[r] * sf + ps;
      acc[0][r] *= sf; acc[1][r] *= sf; acc[2][r] *= sf; acc[3][r] *= sf;
      int row = r + 8 * half;
      lds[row * 32 + nc]      = (bf16)p0;
      lds[row * 32 + 16 + nc] = (bf16)p1;
    }
    // wave-private P staging: DS ops are in-order within a wave, no barrier needed
    const bf16* prow = lds + nc * 32;
    v16bf ap = ld_frag2(prow + half * 8, prow + 16 + half * 8);
#pragma unroll
    for (int ds = 0; ds < 4; ++ds) {
      const bf16* vc = vbuf + (ds * 16 + nc) * 32 + 8 * half;
      v16bf bv_ = ld_frag2(vc, vc + 16);
      acc[ds] = wmma_bf16(ap, bv_, acc[ds]);
    }
    __syncthreads();   // protect kbuf/vbuf before next stage overwrites
  }

  // --- normalize + store O as [B*S, E] bf16 for the vocab GEMM ---
  const size_t obase = ((size_t)bb * SEQ + q0) * E_DIM + h * 64;
#pragma unroll
  for (int r = 0; r < 8; ++r) {
    float inv = 1.0f / l_run[r];
    size_t rowoff = obase + (size_t)(r + 8 * half) * E_DIM;
#pragma unroll
    for (int ds = 0; ds < 4; ++ds)
      Ob[rowoff + ds * 16 + nc] = (bf16)(acc[ds][r] * inv);
  }
}

// ------------- output projection: out[4096,32000] = O @ Wo + bo -------------
// 32x64 register tile per wave (2 A-frags x 4 B-frags, 8 accumulators) -> ~22 flops/B from L2.
// d_out stores are non-temporal so the 500MB stream doesn't evict L2-resident Wo (62.5MB).
__global__ __launch_bounds__(128) void out_gemm(
    const bf16* __restrict__ Ob, const bf16* __restrict__ wot,
    const float* __restrict__ bo, float* __restrict__ out) {
  const int lane = threadIdx.x & 31;
  const int wave = threadIdx.x >> 5;
  const int half = lane >> 4;
  const int nc   = lane & 15;
  const int n0 = blockIdx.x * 64;
  const int m0 = blockIdx.y * 128 + wave * 32;
  const bf16* arow0 = Ob  + (size_t)(m0 + nc) * E_DIM;
  const bf16* arow1 = arow0 + (size_t)16 * E_DIM;
  const bf16* bc    = wot + (size_t)(n0 + nc) * E_DIM + 8 * half;

  v8f acc[2][4] = {};
  for (int k = 0; k < E_DIM; k += 32) {
    v16bf a0 = ld_frag2(arow0 + k + half * 8, arow0 + k + 16 + half * 8);
    v16bf a1 = ld_frag2(arow1 + k + half * 8, arow1 + k + 16 + half * 8);
#pragma unroll
    for (int j = 0; j < 4; ++j) {
      const bf16* bp = bc + (size_t)j * 16 * E_DIM + k;
      v16bf b = ld_frag2(bp, bp + 16);
      acc[0][j] = wmma_bf16(a0, b, acc[0][j]);
      acc[1][j] = wmma_bf16(a1, b, acc[1][j]);
    }
  }
#pragma unroll
  for (int j = 0; j < 4; ++j) {
    float bi = bo[n0 + j * 16 + nc];
#pragma unroll
    for (int i = 0; i < 2; ++i)
#pragma unroll
      for (int r = 0; r < 8; ++r)
        __builtin_nontemporal_store(
            acc[i][j][r] + bi,
            &out[(size_t)(m0 + i * 16 + r + 8 * half) * VOC + n0 + j * 16 + nc]);
  }
}

extern "C" void kernel_launch(void* const* d_in, const int* in_sizes, int n_in,
                              void* d_out, int out_size, void* d_ws, size_t ws_size,
                              hipStream_t stream) {
  (void)in_sizes; (void)n_in; (void)out_size;
  const float* x  = (const float*)d_in[0];
  const float* Wq = (const float*)d_in[1];
  const float* bq = (const float*)d_in[2];
  const float* Wk = (const float*)d_in[3];
  const float* bk = (const float*)d_in[4];
  const float* Wv = (const float*)d_in[5];
  const float* bv = (const float*)d_in[6];
  const float* Wo = (const float*)d_in[7];
  const float* bo = (const float*)d_in[8];
  float* out = (float*)d_out;

  char* wsb = (char*)d_ws;
  size_t off = 0;
  auto alloc = [&](size_t bytes) -> char* {
    char* p = wsb + off;
    off += (bytes + 255) & ~(size_t)255;
    return p;
  };
  bf16* xb  = (bf16*)alloc((size_t)MROWS * E_DIM * 2);
  bf16* wqt = (bf16*)alloc((size_t)E_DIM * E_DIM * 2);
  bf16* wkt = (bf16*)alloc((size_t)E_DIM * E_DIM * 2);
  bf16* wvt = (bf16*)alloc((size_t)E_DIM * E_DIM * 2);
  bf16* wot = (bf16*)alloc((size_t)VOC * E_DIM * 2);
  bf16* Qb  = (bf16*)alloc((size_t)MROWS * E_DIM * 2);
  bf16* Kb  = (bf16*)alloc((size_t)MROWS * E_DIM * 2);
  bf16* Vt  = (bf16*)alloc((size_t)MROWS * E_DIM * 2);
  bf16* Ob  = (bf16*)alloc((size_t)MROWS * E_DIM * 2);
  if (off > ws_size) return;  // workspace too small: deterministic no-op

  cast_f32_bf16<<<(MROWS * E_DIM / 4 + 255) / 256, 256, 0, stream>>>(x, xb, MROWS * E_DIM / 4);
  dim3 tb(32, 8);
  transpose_cast<<<dim3(E_DIM / 32, E_DIM / 32), tb, 0, stream>>>(Wq, wqt, E_DIM, E_DIM);
  transpose_cast<<<dim3(E_DIM / 32, E_DIM / 32), tb, 0, stream>>>(Wk, wkt, E_DIM, E_DIM);
  transpose_cast<<<dim3(E_DIM / 32, E_DIM / 32), tb, 0, stream>>>(Wv, wvt, E_DIM, E_DIM);
  transpose_cast<<<dim3(VOC / 32,  E_DIM / 32), tb, 0, stream>>>(Wo, wot, E_DIM, VOC);
  qkv_gemm<<<dim3(E_DIM / 16, MROWS / 64, 3), 128, 0, stream>>>(
      xb, wqt, wkt, wvt, bq, bk, bv, Qb, Kb, Vt);
  flash_attn<<<BATCH * NHEAD * (SEQ / 64), 128, 0, stream>>>(Qb, Kb, Vt, Ob);
  out_gemm<<<dim3(VOC / 64, MROWS / 128), 128, 0, stream>>>(Ob, wot, bo, out);
}
